// ExpertModule_38053410242743
// MI455X (gfx1250) — matrix operand order repaired and checked by reading, not compiled, not run on hardware
//
#include <hip/hip_runtime.h>

// ---------------- problem constants (from reference) ----------------
#define Bsz   16384     // batch
#define Hdim  512       // hidden
#define Eexp  5         // experts per bank
#define OUTW  128       // expert output width
#define BM    64        // batch rows per workgroup
#define XP    520       // x/h LDS pitch in halves (multiple of 8 -> 16B rows, conflict-free A reads)
#define WP    72        // W-panel LDS pitch in halves (multiple of 8 -> 16B rows)
#define NTHREADS 256    // 8 waves of 32

typedef __attribute__((ext_vector_type(16))) _Float16 v16h;
typedef __attribute__((ext_vector_type(8)))  _Float16 h8;
typedef __attribute__((ext_vector_type(4)))  _Float16 h4;
typedef __attribute__((ext_vector_type(8)))  float    v8f;
typedef __attribute__((ext_vector_type(4)))  float    f4;

// ---- A fragment: 16x32 f16, documented CDNA5 layout ----
// lane = 16*g + m (g = lane>>4); components 0..7 hold K = k0+8g .. k0+8g+7,
// components 8..15 hold K = k0+16+8g .. k0+16+8g+7. Two 16B LDS reads.
__device__ inline v16h lds_load_a(const _Float16* src, int m, int k0, int g) {
  union { v16h v; h8 h[2]; } u;
  const _Float16* p = src + (size_t)m * XP + k0 + 8 * g;
  u.h[0] = *(const h8*)p;
  u.h[1] = *(const h8*)(p + 16);
  return u.v;
}

// ---- B fragment: 32x16 f16 ----
// lane = 16*g + n holds column N = n0+n; components 0..7 hold K = k0+8g .. k0+8g+7,
// components 8..15 hold K = k0+16+8g .. k0+16+8g+7. Strided scalar LDS reads.
__device__ inline v16h lds_load_b(const _Float16* wsd, int k0, int lane, int n0) {
  v16h v;
  const _Float16* p = wsd + (size_t)(k0 + 8 * (lane >> 4)) * WP + n0 + (lane & 15);
  #pragma unroll
  for (int s = 0; s < 8; ++s) {
    v[s]     = p[(size_t)s * WP];
    v[8 + s] = p[(size_t)(16 + s) * WP];
  }
  return v;
}

// Cooperative fp32->f16 stage of the 64x512 x tile into LDS.
__device__ inline void stage_x(const float* __restrict__ xg, int rowBase,
                               _Float16* xs, int tid) {
  #pragma unroll
  for (int i = 0; i < (BM * Hdim / 4) / NTHREADS; ++i) {   // 32 iters
    int lin = i * NTHREADS + tid;
    int row = lin >> 7;          // 128 float4 per row
    int c4  = lin & 127;
    f4 v = *(const f4*)(xg + (size_t)(rowBase + row) * Hdim + 4 * c4);
    h4 h;
    h.x = (_Float16)v.x; h.y = (_Float16)v.y;
    h.z = (_Float16)v.z; h.w = (_Float16)v.w;
    *(h4*)(xs + (size_t)row * XP + 4 * c4) = h;
  }
}

// Cooperative fp32->f16 stage of a 512x64 weight panel (cols c0..c0+63) into LDS.
__device__ inline void stage_w(const float* __restrict__ Wg, int ld, int c0,
                               _Float16* wsd, int tid) {
  #pragma unroll
  for (int i = 0; i < (Hdim * 64 / 4) / NTHREADS; ++i) {   // 32 iters
    int lin = i * NTHREADS + tid;
    int row = lin >> 4;          // 16 float4 per panel row
    int c4  = lin & 15;
    f4 v = *(const f4*)(Wg + (size_t)row * ld + c0 + 4 * c4);
    h4 h;
    h.x = (_Float16)v.x; h.y = (_Float16)v.y;
    h.z = (_Float16)v.z; h.w = (_Float16)v.w;
    *(h4*)(wsd + (size_t)row * WP + 4 * c4) = h;
  }
}

// One 16x32 output strip per wave of a 64x64 panel; K = 512 over f16 WMMA (K=32).
__device__ inline void gemm_panel(const _Float16* __restrict__ src,
                                  const _Float16* __restrict__ wsd,
                                  int m0, int ml, int g, int lane, int nloc,
                                  v8f& acc0, v8f& acc1) {
  acc0 = (v8f){0.f,0.f,0.f,0.f,0.f,0.f,0.f,0.f};
  acc1 = acc0;
  #pragma unroll 4
  for (int k0 = 0; k0 < Hdim; k0 += 32) {
    v16h a  = lds_load_a(src, m0 + ml, k0, g);
    v16h b0 = lds_load_b(wsd, k0, lane, nloc);
    v16h b1 = lds_load_b(wsd, k0, lane, nloc + 16);
    acc0 = __builtin_amdgcn_wmma_f32_16x16x32_f16(false, a, false, b0,
                                                  (short)0, acc0, false, false);
    acc1 = __builtin_amdgcn_wmma_f32_16x16x32_f16(false, a, false, b1,
                                                  (short)0, acc1, false, false);
  }
}

__global__ __launch_bounds__(NTHREADS)
void moe_expert_kernel(const float* __restrict__ share_x,
                       const float* __restrict__ tx0,
                       const float* __restrict__ tx1,
                       const float* __restrict__ sW1, const float* __restrict__ sb1,
                       const float* __restrict__ sW2, const float* __restrict__ sb2,
                       const float* __restrict__ tW1, const float* __restrict__ tb1,
                       const float* __restrict__ tW2, const float* __restrict__ tb2,
                       float* __restrict__ outp) {
  extern __shared__ _Float16 smem[];
  _Float16* xs = smem;                          // [64][520] f16  x tile
  _Float16* hs = smem + (size_t)BM * XP;        // [64][520] f16  relu(x@W1+b1)
  _Float16* ws = smem + (size_t)2 * BM * XP;    // [512][72] f16  weight panel

  const int tid  = threadIdx.x;
  const int lane = tid & 31;
  const int wave = tid >> 5;
  const int m0   = 16 * (wave & 3);   // wave's row block within the 64-row tile
  const int nloc = 32 * (wave >> 2);  // wave's 32-col strip within the 64-col panel
  const int g    = lane >> 4;
  const int ml   = lane & 15;

  const int tileIdx = blockIdx.x;     // 0..255
  const int e       = blockIdx.y;     // expert
  const int bank    = blockIdx.z;     // 0=share, 1=task0, 2=task1
  const int rowBase = tileIdx * BM;

  const float *xg, *W1p, *b1p, *W2p, *b2p;
  if (bank == 0) {
    xg = share_x;
    W1p = sW1 + (size_t)e * Hdim * Hdim;  b1p = sb1 + (size_t)e * Hdim;
    W2p = sW2 + (size_t)e * Hdim * OUTW;  b2p = sb2 + (size_t)e * OUTW;
  } else {
    int t = bank - 1;
    xg = t ? tx1 : tx0;
    size_t te = (size_t)t * Eexp + e;
    W1p = tW1 + te * Hdim * Hdim;  b1p = tb1 + te * Hdim;
    W2p = tW2 + te * Hdim * OUTW;  b2p = tb2 + te * OUTW;
  }
  const size_t outBase = ((size_t)bank * Eexp + e) * (size_t)Bsz * OUTW;

  // ---------------- phase 0: stage x tile ----------------
  stage_x(xg, rowBase, xs, tid);
  __syncthreads();

  // ---------------- phase 1: h = relu(x @ W1 + b1), 8 panels of 64 cols ----------------
  for (int p = 0; p < Hdim / 64; ++p) {
    stage_w(W1p, Hdim, p * 64, ws, tid);
    if (p + 1 < Hdim / 64)
      __builtin_prefetch(W1p + (size_t)(p + 1) * 64, 0, 1);  // global_prefetch_b8
    __syncthreads();

    v8f acc0, acc1;
    gemm_panel(xs, ws, m0, ml, g, lane, nloc, acc0, acc1);

    const int ncol = p * 64 + nloc;            // column within H
    const float bia0 = b1p[ncol + ml];
    const float bia1 = b1p[ncol + 16 + ml];
    #pragma unroll
    for (int r = 0; r < 8; ++r) {
      int m = m0 + r + 8 * g;                  // C/D layout: lanes 16-31 -> M+8
      float v0 = acc0[r] + bia0; v0 = v0 > 0.f ? v0 : 0.f;
      float v1 = acc1[r] + bia1; v1 = v1 > 0.f ? v1 : 0.f;
      hs[(size_t)m * XP + ncol + ml]      = (_Float16)v0;
      hs[(size_t)m * XP + ncol + 16 + ml] = (_Float16)v1;
    }
    __syncthreads();   // hs writes visible / ws safe to overwrite
  }

  // ---------------- phase 2: out = h @ W2 + b2, 2 panels of 64 cols ----------------
  for (int p = 0; p < OUTW / 64; ++p) {
    stage_w(W2p, OUTW, p * 64, ws, tid);
    __syncthreads();

    v8f acc0, acc1;
    gemm_panel(hs, ws, m0, ml, g, lane, nloc, acc0, acc1);

    const int ncol = p * 64 + nloc;            // column within OUT
    const float bia0 = b2p[ncol + ml];
    const float bia1 = b2p[ncol + 16 + ml];
    #pragma unroll
    for (int r = 0; r < 8; ++r) {
      int m = m0 + r + 8 * g;
      size_t ro = outBase + (size_t)(rowBase + m) * OUTW + ncol;
      outp[ro + ml]      = acc0[r] + bia0;
      outp[ro + 16 + ml] = acc1[r] + bia1;
    }
    __syncthreads();
  }
}

extern "C" void kernel_launch(void* const* d_in, const int* in_sizes, int n_in,
                              void* d_out, int out_size, void* d_ws, size_t ws_size,
                              hipStream_t stream) {
  const float* share_x = (const float*)d_in[0];
  const float* tx0     = (const float*)d_in[1];
  const float* tx1     = (const float*)d_in[2];
  const float* sW1     = (const float*)d_in[3];
  const float* sb1     = (const float*)d_in[4];
  const float* sW2     = (const float*)d_in[5];
  const float* sb2     = (const float*)d_in[6];
  const float* tW1     = (const float*)d_in[7];
  const float* tb1     = (const float*)d_in[8];
  const float* tW2     = (const float*)d_in[9];
  const float* tb2     = (const float*)d_in[10];
  float* outp          = (float*)d_out;

  dim3 grid(Bsz / BM, Eexp, 3);
  dim3 block(NTHREADS);
  size_t shmem = ((size_t)2 * BM * XP + (size_t)Hdim * WP) * 2;  // f16 bytes = 206848

  hipLaunchKernelGGL(moe_expert_kernel, grid, block, shmem, stream,
                     share_x, tx0, tx1, sW1, sb1, sW2, sb2,
                     tW1, tb1, tW2, tb2, outp);
}
